// Attention_60748017434692
// MI455X (gfx1250) — matrix operand order repaired and checked by reading, not compiled, not run on hardware
//
#include <hip/hip_runtime.h>
#include <math.h>
#include <type_traits>

typedef _Float16 v16h __attribute__((ext_vector_type(16)));
typedef _Float16 v8h  __attribute__((ext_vector_type(8)));
typedef float    v8f  __attribute__((ext_vector_type(8)));

#define ATT_B 16
#define ATT_L 2048
#define ATT_D 64
#define WAVES_PER_BLOCK 4
#define QROWS 32           // query rows per wave: 2 stripes of 16
#define SROW 36            // padded LDS row stride (floats): conflict-free + 16B aligned
#define LDS_PER_WAVE (2*16*SROW + 2*32)   // 2 stage tiles + 2 per-stripe xslot regions

// ================= pre-pass kernels: K -> f16, V -> f16 transposed =================

__global__ __launch_bounds__(256)
void convert_k_f16(const float* __restrict__ in, _Float16* __restrict__ out) {
    const size_t i0 = ((size_t)blockIdx.x * 256 + threadIdx.x) * 8;
    v8h h;
#pragma unroll
    for (int j = 0; j < 8; ++j) h[j] = (_Float16)in[i0 + j];
    *(v8h*)(out + i0) = h;
}

// Vt[b][d][k] = (f16) V[b][k][d]; LDS-tiled so both phases are coalesced
__global__ __launch_bounds__(256)
void transpose_v_f16(const float* __restrict__ V, _Float16* __restrict__ Vt) {
    __shared__ _Float16 tile[ATT_D * 33];
    const int k0 = blockIdx.x * 32;
    const int b  = blockIdx.y;
    const float*    Vb  = V  + (size_t)b * ATT_L * ATT_D;
    _Float16*       Vtb = Vt + (size_t)b * ATT_D * ATT_L;
    {
        const int d   = threadIdx.x & 63;
        const int kk0 = threadIdx.x >> 6;   // 0..3
#pragma unroll
        for (int r = 0; r < 8; ++r) {
            const int kk = kk0 + r * 4;
            tile[d * 33 + kk] = (_Float16)Vb[(size_t)(k0 + kk) * ATT_D + d];
        }
    }
    __syncthreads();
    {
        const int kk = threadIdx.x & 31;
        const int d0 = threadIdx.x >> 5;    // 0..7
#pragma unroll
        for (int r = 0; r < 8; ++r) {
            const int d = d0 + r * 8;
            Vtb[(size_t)d * ATT_L + k0 + kk] = tile[d * 33 + kk];
        }
    }
}

// ================= fragment loaders =================

// A-matrix 16x32 f16 layout (contiguous k order): lane<16 holds row=lane, K = 0..15;
// lane>=16 holds row=lane-16, K = 16..31.  (hi = lane>>4)
// `mul` is applied during conversion (exact for powers of two).
__device__ __forceinline__ v16h load_a_frag_f32(const float* __restrict__ base,
                                                int row, int d0, int hi, float mul) {
    const float* p = base + (size_t)row * ATT_D + d0 + hi * 16;
    v16h r;
#pragma unroll
    for (int j = 0; j < 8; ++j) r[j] = (_Float16)(p[j] * mul);
#pragma unroll
    for (int j = 0; j < 8; ++j) r[8 + j] = (_Float16)(p[8 + j] * mul);
    return r;
}

// B fragment (32x16): lane = N column; lanes 0-15 hold K=0..15, lanes 16-31 K=16..31.
// f32 fallback versions:
__device__ __forceinline__ v16h load_b_frag_kT_f32(const float* __restrict__ base,
                                                   int keyrow, int d0, int hi) {
    const float* p = base + (size_t)keyrow * ATT_D + d0 + hi * 16;
    v16h r;
#pragma unroll
    for (int j = 0; j < 16; ++j) r[j] = (_Float16)p[j];
    return r;
}
__device__ __forceinline__ v16h load_b_frag_v_f32(const float* __restrict__ base,
                                                  int kb, int dcol, int hi) {
    const float* p = base + (size_t)(kb + hi * 16) * ATT_D + dcol;
    v16h r;
#pragma unroll
    for (int j = 0; j < 16; ++j) r[j] = (_Float16)p[j * ATT_D];
    return r;
}

// ================= main flash-attention kernel =================

template <bool F16PRE>
__global__ __launch_bounds__(32 * WAVES_PER_BLOCK)
void fa_fwd_causal_wmma(const float* __restrict__ Q,
                        const float* __restrict__ Kf,
                        const float* __restrict__ Vf,
                        const _Float16* __restrict__ Kh,
                        const _Float16* __restrict__ Vt,
                        float* __restrict__ O) {
    __shared__ float lds[WAVES_PER_BLOCK * LDS_PER_WAVE];

    const int lane = threadIdx.x & 31;
    const int wave = threadIdx.x >> 5;
    const int hi   = lane >> 4;
    const int lo   = lane & 15;

    const int t  = blockIdx.x * WAVES_PER_BLOCK + wave; // 32-row tile index, 0..63
    const int b  = blockIdx.y;
    const int q0 = t * QROWS;

    const float*    Qb  = Q  + (size_t)b * ATT_L * ATT_D;
    const float*    Kfb = Kf + (size_t)b * ATT_L * ATT_D;
    const float*    Vfb = Vf + (size_t)b * ATT_L * ATT_D;
    const _Float16* Khb = Kh + (size_t)b * ATT_L * ATT_D;
    const _Float16* Vtb = Vt + (size_t)b * ATT_D * ATT_L;
    float*          Ob  = O  + (size_t)b * ATT_L * ATT_D;

    float* myL   = lds + wave * LDS_PER_WAVE;
    float* xbase = myL + 2 * 16 * SROW;     // 2 x 32 floats: per-stripe partial-max slots

    auto fetch_kT = [&](int keyrow, int d0) -> v16h {
        if constexpr (F16PRE)
            return *(const v16h*)(Khb + (size_t)keyrow * ATT_D + d0 + hi * 16);
        else
            return load_b_frag_kT_f32(Kfb, keyrow, d0, hi);
    };
    auto fetch_v = [&](int kb, int dcol) -> v16h {
        if constexpr (F16PRE)
            return *(const v16h*)(Vtb + (size_t)dcol * ATT_L + kb + hi * 16);
        else
            return load_b_frag_v_f32(Vfb, kb, dcol, hi);
    };

    // Q fragments, pre-scaled by 1/sqrt(D) = 0.125 (power of two -> exact in f16)
    v16h qf[2][2];
#pragma unroll
    for (int s = 0; s < 2; ++s) {
        qf[s][0] = load_a_frag_f32(Qb, q0 + 16 * s + lo, 0,  hi, 0.125f);
        qf[s][1] = load_a_frag_f32(Qb, q0 + 16 * s + lo, 32, hi, 0.125f);
    }

    // ones B-fragment for row-sum WMMA (P x 1 -> row sums in C/D layout)
    v16h ones;
#pragma unroll
    for (int j = 0; j < 16; ++j) ones[j] = (_Float16)1.0f;

    v8f acc[2][4] = {};
    float l[2][8];
    float m_t[2];         // row max, transposed layout (row = lo)
    float m_cd[2][8];     // row max, C/D layout (row = i + 8*hi) — same values
#pragma unroll
    for (int s = 0; s < 2; ++s) {
        m_t[s] = -INFINITY;
#pragma unroll
        for (int i = 0; i < 8; ++i) { l[s][i] = 0.0f; m_cd[s][i] = -INFINITY; }
    }

    auto chunk = [&](int kb, auto maskc) {
        constexpr bool MASK = decltype(maskc)::value;

        // K fragments for this 32-key chunk, shared by both stripes
        const v16h kf00 = fetch_kT(kb + lo,      0);
        const v16h kf01 = fetch_kT(kb + lo,      32);
        const v16h kf10 = fetch_kT(kb + 16 + lo, 0);
        const v16h kf11 = fetch_kT(kb + 16 + lo, 32);

        // branchless prefetch of the next chunk (clamped: rows <= q0+31 < L)
        if constexpr (F16PRE) {
            const int pk = (kb + 32 <= q0) ? (kb + 32) : q0;
            __builtin_prefetch(Khb + (size_t)(pk + lane) * ATT_D, 0, 3);
            __builtin_prefetch(Vtb + (size_t)lane        * ATT_L + pk, 0, 3);
            __builtin_prefetch(Vtb + (size_t)(lane + 32) * ATT_L + pk, 0, 3);
        }

        v16h pf[2];

#pragma unroll
        for (int s = 0; s < 2; ++s) {
            // ---- S = Q * K^T for this stripe (C/D layout, already scaled) ----
            v8f s0 = {}, s1 = {};
            s0 = __builtin_amdgcn_wmma_f32_16x16x32_f16(false, qf[s][0], false, kf00, (short)0, s0, false, false);
            s0 = __builtin_amdgcn_wmma_f32_16x16x32_f16(false, qf[s][1], false, kf01, (short)0, s0, false, false);
            s1 = __builtin_amdgcn_wmma_f32_16x16x32_f16(false, qf[s][0], false, kf10, (short)0, s1, false, false);
            s1 = __builtin_amdgcn_wmma_f32_16x16x32_f16(false, qf[s][1], false, kf11, (short)0, s1, false, false);

            // ---- (causal mask on diagonal chunk), stage to LDS ----
            float* stage = myL + s * (16 * SROW);
#pragma unroll
            for (int i = 0; i < 8; ++i) {
                float v0 = s0[i];
                float v1 = s1[i];
                if constexpr (MASK) {
                    const int qi = q0 + 16 * s + i + 8 * hi;
                    if (kb + lo      > qi) v0 = -1e30f;
                    if (kb + 16 + lo > qi) v1 = -1e30f;
                }
                const int row = i + 8 * hi;
                stage[row * SROW + lo]      = v0;
                stage[row * SROW + 16 + lo] = v1;
            }
            asm volatile("" ::: "memory");  // LDS is in-order within a wave

            // ---- transposed read: lane owns row `lo`, its A-fragment K-slice ----
            float tv[16];
            {
                const float* rp = stage + lo * SROW + 16 * hi;
#pragma unroll
                for (int j = 0; j < 8; ++j) { tv[j] = rp[j]; tv[8 + j] = rp[8 + j]; }
            }

            // ---- row max: lane-local tree + ONE cross-half LDS exchange ----
            float pm = tv[0];
#pragma unroll
            for (int j = 1; j < 16; ++j) pm = fmaxf(pm, tv[j]);
            float* xs = xbase + s * 32;     // per-stripe region (read again below)
            xs[lane] = pm;
            asm volatile("" ::: "memory");
            const float rowmax = fmaxf(pm, xs[lane ^ 16]);

            const float mn = fmaxf(m_t[s], rowmax);
            m_t[s] = mn;

            // ---- build P fragment directly in registers (A layout) ----
#pragma unroll
            for (int j = 0; j < 16; ++j) pf[s][j] = (_Float16)__expf(tv[j] - mn);
        }
        asm volatile("" ::: "memory");

        // ---- C/D side: recompute mn/alpha from the staged partials, update l/acc ----
#pragma unroll
        for (int s = 0; s < 2; ++s) {
            const float* xs = xbase + s * 32;
            float al[8];
#pragma unroll
            for (int i = 0; i < 8; ++i) {
                const float rmax = fmaxf(xs[i + 8 * hi], xs[i + 8 * hi + 16]);
                const float mn   = fmaxf(m_cd[s][i], rmax);
                al[i] = __expf(m_cd[s][i] - mn);
                m_cd[s][i] = mn;
            }

            // row sums of P via WMMA against the ones matrix (C/D layout result)
            v8f rs = {};
            rs = __builtin_amdgcn_wmma_f32_16x16x32_f16(false, pf[s], false, ones, (short)0, rs, false, false);

#pragma unroll
            for (int i = 0; i < 8; ++i) {
                l[s][i] = l[s][i] * al[i] + rs[i];
                acc[s][0][i] *= al[i]; acc[s][1][i] *= al[i];
                acc[s][2][i] *= al[i]; acc[s][3][i] *= al[i];
            }
        }

        // ---- O += P * V : V fragment loaded once, used by both stripes ----
#pragma unroll
        for (int g = 0; g < 4; ++g) {
            const v16h bv = fetch_v(kb, g * 16 + lo);
            acc[0][g] = __builtin_amdgcn_wmma_f32_16x16x32_f16(false, pf[0], false, bv, (short)0, acc[0][g], false, false);
            acc[1][g] = __builtin_amdgcn_wmma_f32_16x16x32_f16(false, pf[1], false, bv, (short)0, acc[1][g], false, false);
        }
    };

    // keys 0..q0+31 needed; all chunks before the diagonal one are unmasked
    const int nfull = q0 >> 5;
    for (int c = 0; c < nfull; ++c) chunk(c * 32, std::false_type{});
    chunk(q0, std::true_type{}); // diagonal chunk, causal-masked

    // ---------- epilogue: normalize and store (coalesced b32 across lanes) ----------
#pragma unroll
    for (int s = 0; s < 2; ++s)
#pragma unroll
        for (int i = 0; i < 8; ++i) {
            const float inv = 1.0f / l[s][i];
            float* orow = Ob + (size_t)(q0 + 16 * s + i + 8 * hi) * ATT_D + lo;
            orow[0]  = acc[s][0][i] * inv;
            orow[16] = acc[s][1][i] * inv;
            orow[32] = acc[s][2][i] * inv;
            orow[48] = acc[s][3][i] * inv;
        }
}

extern "C" void kernel_launch(void* const* d_in, const int* in_sizes, int n_in,
                              void* d_out, int out_size, void* d_ws, size_t ws_size,
                              hipStream_t stream) {
    (void)in_sizes; (void)n_in; (void)out_size;
    const float* Q = (const float*)d_in[0];
    const float* K = (const float*)d_in[1];
    const float* V = (const float*)d_in[2];
    float*       O = (float*)d_out;

    const size_t elems    = (size_t)ATT_B * ATT_L * ATT_D;
    const size_t f16bytes = elems * sizeof(_Float16);

    dim3 grid(ATT_L / QROWS / WAVES_PER_BLOCK, ATT_B); // (16, 16)
    dim3 block(32 * WAVES_PER_BLOCK);                  // 4 waves

    if (ws_size >= 2 * f16bytes) {
        _Float16* Kh = (_Float16*)d_ws;
        _Float16* Vt = (_Float16*)((char*)d_ws + f16bytes);
        hipLaunchKernelGGL(convert_k_f16, dim3(elems / (256 * 8)), dim3(256), 0, stream, K, Kh);
        hipLaunchKernelGGL(transpose_v_f16, dim3(ATT_L / 32, ATT_B), dim3(256), 0, stream, V, Vt);
        hipLaunchKernelGGL((fa_fwd_causal_wmma<true>), grid, block, 0, stream, Q, K, V, Kh, Vt, O);
    } else {
        hipLaunchKernelGGL((fa_fwd_causal_wmma<false>), grid, block, 0, stream,
                           Q, K, V, (const _Float16*)nullptr, (const _Float16*)nullptr, O);
    }
}
